// MaskedAttention_5643587027407
// MI455X (gfx1250) — matrix operand order repaired and checked by reading, not compiled, not run on hardware
//
#include <hip/hip_runtime.h>
#include <hip/hip_bf16.h>

typedef __attribute__((ext_vector_type(16))) _Float16     v16h;
typedef __attribute__((ext_vector_type(8)))  float        v8f;
typedef __attribute__((ext_vector_type(4)))  unsigned int u32x4;
typedef __attribute__((ext_vector_type(8)))  int          i32x8;
typedef __attribute__((ext_vector_type(4)))  int          i32x4;

#define SEQ   2048
#define DIM   768
#define HEADS 12
#define DK    64
#define BATCH 4
#define MROWS (BATCH * SEQ)          // 8192
#define NEG_BIG (-3.0e38f)

// GEMM block tiling
#define BM 128
#define BN 64
#define BK 32

// ---------------------------------------------------------------------------
// Fragment loaders (wave32, 16x16x32 f16 WMMA; same contiguous k order for A and B)
// ---------------------------------------------------------------------------

// A fragment: 16x32 f16 tile from row-major A (ld = lda), origin (row0, k0).
// lane L: row = row0 + (L&15); elems e -> K = k0 + half*16 + e (16 contiguous halfwords).
__device__ inline v16h load_a_frag(const _Float16* A, int lda, int row0, int k0) {
    const int lane = threadIdx.x & 31;
    const int half = lane >> 4;
    const _Float16* p = A + (size_t)(row0 + (lane & 15)) * lda + k0 + half * 16;
    v16h a;
#pragma unroll
    for (int e = 0; e < 8; ++e) a[e] = p[e];
#pragma unroll
    for (int e = 0; e < 8; ++e) a[8 + e] = p[8 + e];
    return a;
}

// B fragment where B[k][n] = W[n0+n][k] (multiply by W^T), W row-major ld=ldw.
// lane L: n = L&15; elems e -> K = k0 + half*16 + e (16 contiguous halfwords).
__device__ inline v16h load_b_frag_rowT(const _Float16* W, int ldw, int n0, int k0) {
    const int lane = threadIdx.x & 31;
    const int half = lane >> 4;
    const _Float16* p = W + (size_t)(n0 + (lane & 15)) * ldw + k0 + half * 16;
    v16h b;
#pragma unroll
    for (int e = 0; e < 16; ++e) b[e] = p[e];
    return b;
}

// ---------------------------------------------------------------------------
// Tensor Data Mover: 2D tile (tile_d0 x tile_d1, f16) from global -> LDS.
// Descriptor per CDNA5 ISA §8.3/8.4. Issued per-wave; tracked by TENSORcnt.
// ---------------------------------------------------------------------------
__device__ inline void tdm_load_tile_f16(unsigned lds_byte_addr, const void* gtile,
                                         unsigned tile_d0, unsigned tile_d1,
                                         unsigned long long tensor_d0,
                                         unsigned long long tensor_d1,
                                         unsigned long long stride0_elems) {
    const unsigned long long ga = (unsigned long long)(size_t)gtile;
    u32x4 g0;
    g0[0] = 1u;                                         // count=1, user descriptor
    g0[1] = lds_byte_addr;                              // LDS dest (bytes)
    g0[2] = (unsigned)(ga & 0xFFFFFFFFu);               // global_addr[31:0]
    g0[3] = (unsigned)((ga >> 32) & 0x01FFFFFFu)        // global_addr[56:32]
          | (2u << 30);                                 // type=2 ("image")
    // group1 (256 bits): data_size=1 (2 bytes); no mask/barrier/pad/iterate
    const unsigned dw0 = (1u << 16);
    const unsigned dw1 = (unsigned)((tensor_d0 & 0xFFFFu) << 16);          // dim0[15:0]
    const unsigned dw2 = (unsigned)((tensor_d0 >> 16) & 0xFFFFu)           // dim0[31:16]
                       | (unsigned)((tensor_d1 & 0xFFFFu) << 16);          // dim1[15:0]
    const unsigned dw3 = (unsigned)((tensor_d1 >> 16) & 0xFFFFu)           // dim1[31:16]
                       | ((tile_d0 & 0xFFFFu) << 16);                      // tile_dim0
    const unsigned dw4 = (tile_d1 & 0xFFFFu);                              // tile_dim1 (dim2=0)
    const unsigned dw5 = (unsigned)(stride0_elems & 0xFFFFFFFFu);          // stride0[31:0]
    const unsigned dw6 = (unsigned)((stride0_elems >> 32) & 0xFFFFu);      // stride0[47:32]
    const unsigned dw7 = 0u;                                               // stride1 unused (2D)
    i32x8 g1 = { (int)dw0, (int)dw1, (int)dw2, (int)dw3,
                 (int)dw4, (int)dw5, (int)dw6, (int)dw7 };
    i32x4 z4 = { 0, 0, 0, 0 };
#if __clang_major__ >= 23
    i32x8 z8 = { 0, 0, 0, 0, 0, 0, 0, 0 };
    __builtin_amdgcn_tensor_load_to_lds(g0, g1, z4, z4, z8, 0);
#else
    __builtin_amdgcn_tensor_load_to_lds(g0, g1, z4, z4, 0);
#endif
}

// ---------------------------------------------------------------------------
// Kernel 0: f32 -> f16 conversion
// ---------------------------------------------------------------------------
__global__ void cvt_f32_f16(const float* __restrict__ in, _Float16* __restrict__ out, int n) {
    int i = blockIdx.x * blockDim.x + threadIdx.x;
    int stride = gridDim.x * blockDim.x;
    for (; i < n; i += stride) out[i] = (_Float16)in[i];
}

// ---------------------------------------------------------------------------
// Tiled GEMM: OUT = X (MROWS x DIM) @ W^T + bias, X/W f16, f32 accumulate.
// Block = 256 threads (8 waves), tile BM x BN, K-loop in BK slices.
// LDS double-buffered via TDM (wave 0 issues; TENSORcnt + barrier publishes).
// Each wave owns a 32x32 register tile (4 WMMA accumulators).
// MODE=1: f16 scatter to [b,h,s,dk]   (Q, K)
// MODE=2: f16 scatter to [b,h,dk,s]   (V transposed; contiguous s-stores)
// MODE=0: f32 row-major [MROWS,DIM]   (output projection)
// grid = (DIM/BN, MROWS/BM)
// ---------------------------------------------------------------------------
template <int MODE>
__global__ void gemm_tiled(const _Float16* __restrict__ X, const _Float16* __restrict__ W,
                           const float* __restrict__ bias,
                           _Float16* __restrict__ O16, float* __restrict__ O32) {
    __shared__ _Float16 sA[2][BM][BK];   // 16 KB
    __shared__ _Float16 sB[2][BN][BK];   //  8 KB
    const int n0g  = blockIdx.x * BN;
    const int m0g  = blockIdx.y * BM;
    const int wave = threadIdx.x >> 5;   // 0..7
    const int lane = threadIdx.x & 31;
    const int half = lane >> 4;
    const int wm   = wave >> 1;          // 0..3 (M)
    const int wn   = wave & 1;           // 0..1 (N)

    // Prologue: stage K-slice 0 into buffer 0.
    if (wave == 0) {
        tdm_load_tile_f16((unsigned)(size_t)&sA[0][0][0], X + (size_t)m0g * DIM,
                          BK, BM, DIM, MROWS, DIM);
        tdm_load_tile_f16((unsigned)(size_t)&sB[0][0][0], W + (size_t)n0g * DIM,
                          BK, BN, DIM, DIM, DIM);
    }

    v8f acc00 = {}, acc01 = {}, acc10 = {}, acc11 = {};

    for (int kb = 0; kb < DIM / BK; ++kb) {
        const int buf = kb & 1;
        if (wave == 0) __builtin_amdgcn_s_wait_tensorcnt(0);  // current buffer landed
        __syncthreads();                                      // publish to all waves
        if (wave == 0 && (kb + 1) < DIM / BK) {               // prefetch next slice
            const int k1 = (kb + 1) * BK;
            tdm_load_tile_f16((unsigned)(size_t)&sA[buf ^ 1][0][0],
                              X + (size_t)m0g * DIM + k1, BK, BM, DIM, MROWS, DIM);
            tdm_load_tile_f16((unsigned)(size_t)&sB[buf ^ 1][0][0],
                              W + (size_t)n0g * DIM + k1, BK, BN, DIM, DIM, DIM);
        }
        v16h a0 = load_a_frag(&sA[buf][wm * 32][0],      BK, 0, 0);
        v16h a1 = load_a_frag(&sA[buf][wm * 32 + 16][0], BK, 0, 0);
        v16h b0 = load_b_frag_rowT(&sB[buf][0][0], BK, wn * 32, 0);
        v16h b1 = load_b_frag_rowT(&sB[buf][0][0], BK, wn * 32 + 16, 0);
        acc00 = __builtin_amdgcn_wmma_f32_16x16x32_f16(false, a0, false, b0, (short)0, acc00, false, false);
        acc01 = __builtin_amdgcn_wmma_f32_16x16x32_f16(false, a0, false, b1, (short)0, acc01, false, false);
        acc10 = __builtin_amdgcn_wmma_f32_16x16x32_f16(false, a1, false, b0, (short)0, acc10, false, false);
        acc11 = __builtin_amdgcn_wmma_f32_16x16x32_f16(false, a1, false, b1, (short)0, acc11, false, false);
    }

    // Epilogue: 4 tiles of 16x16 per wave.
#pragma unroll
    for (int mt = 0; mt < 2; ++mt) {
#pragma unroll
        for (int nt = 0; nt < 2; ++nt) {
            const v8f acc = (mt == 0) ? (nt == 0 ? acc00 : acc01)
                                      : (nt == 0 ? acc10 : acc11);
            const int col = n0g + wn * 32 + nt * 16 + (lane & 15);
            const float bv = bias[col];
#pragma unroll
            for (int r = 0; r < 8; ++r) {
                const int m = m0g + wm * 32 + mt * 16 + half * 8 + r;
                if (MODE == 1) {                        // [b,h,s,dk]
                    const int head = col >> 6;
                    const int dkc  = col & (DK - 1);
                    const int bi   = m >> 11;
                    const int sl   = m & (SEQ - 1);
                    O16[(((size_t)bi * HEADS + head) * SEQ + sl) * DK + dkc] =
                        (_Float16)(acc[r] + bv);
                } else if (MODE == 2) {                 // [b,h,dk,s] (V^T)
                    const int head = col >> 6;
                    const int dkc  = col & (DK - 1);
                    const int bi   = m >> 11;
                    const int sl   = m & (SEQ - 1);
                    O16[(((size_t)bi * HEADS + head) * DK + dkc) * SEQ + sl] =
                        (_Float16)(acc[r] + bv);
                } else {                                // f32 row-major
                    O32[(size_t)m * DIM + col] = acc[r] + bv;
                }
            }
        }
    }
}

// ---------------------------------------------------------------------------
// Kernel 2: causal flash attention.
// Q,K : [b,h,s,dk] f16.  Vt : [b,h,dk,s] f16 (transposed).  O: [b,s,d] f16.
// grid = (SEQ/16, BATCH*HEADS), block = 32 (one wave per 16-row q tile)
// ---------------------------------------------------------------------------
__global__ void attn_fwd(const _Float16* __restrict__ Q, const _Float16* __restrict__ K,
                         const _Float16* __restrict__ Vt, _Float16* __restrict__ O) {
    const int bh   = blockIdx.y;
    const int bi   = bh / HEADS;
    const int head = bh % HEADS;
    const size_t base = (size_t)bh * SEQ * DK;
    const _Float16* Qp  = Q  + base;
    const _Float16* Kp  = K  + base;
    const _Float16* Vtp = Vt + base;     // [dk][s], ld = SEQ

    const int q0   = blockIdx.x * 16;
    const int lane = threadIdx.x & 31;
    const int half = lane >> 4;

    __shared__ float    sS[16][32];
    __shared__ _Float16 sP[16][32];
    __shared__ float    sM[16], sL[16], sScale[16];

    if (lane < 16) { sM[lane] = NEG_BIG; sL[lane] = 0.0f; }
    __syncthreads();

    // Q fragments are loop-invariant: load once.
    const v16h qa0 = load_a_frag(Qp, DK, q0, 0);
    const v16h qa1 = load_a_frag(Qp, DK, q0, 32);

    v8f o0 = {}, o1 = {}, o2 = {}, o3 = {};

    for (int kv0 = 0; kv0 < q0 + 16; kv0 += 32) {
        // ---- S = Q tile @ K tile^T (scaled) into LDS ----
#pragma unroll
        for (int nt = 0; nt < 2; ++nt) {
            v8f s = {};
            v16h kb0 = load_b_frag_rowT(Kp, DK, kv0 + nt * 16, 0);
            v16h kb1 = load_b_frag_rowT(Kp, DK, kv0 + nt * 16, 32);
            s = __builtin_amdgcn_wmma_f32_16x16x32_f16(false, qa0, false, kb0, (short)0, s, false, false);
            s = __builtin_amdgcn_wmma_f32_16x16x32_f16(false, qa1, false, kb1, (short)0, s, false, false);
#pragma unroll
            for (int r = 0; r < 8; ++r)
                sS[half * 8 + r][nt * 16 + (lane & 15)] = s[r] * 0.125f;
        }
        __syncthreads();

        // ---- per-row causal mask + online softmax update (one lane per row) ----
        if (lane < 16) {
            const int qi = q0 + lane;
            const float m_old = sM[lane];
            float m_new = m_old;
#pragma unroll
            for (int c = 0; c < 32; ++c) {
                const int ki = kv0 + c;
                float v = (ki <= qi) ? sS[lane][c] : NEG_BIG;
                sS[lane][c] = v;
                m_new = fmaxf(m_new, v);
            }
            const float corr = __expf(m_old - m_new);
            float l = sL[lane] * corr;
#pragma unroll
            for (int c = 0; c < 32; ++c) {
                const float p = __expf(sS[lane][c] - m_new);
                sP[lane][c] = (_Float16)p;
                l += p;
            }
            sM[lane] = m_new;
            sL[lane] = l;
            sScale[lane] = corr;
        }
        __syncthreads();

        // ---- rescale running O, then O += P @ V-tile ----
#pragma unroll
        for (int r = 0; r < 8; ++r) {
            const float c = sScale[half * 8 + r];
            o0[r] *= c; o1[r] *= c; o2[r] *= c; o3[r] *= c;
        }
        // B[k][n] = V[kv0+k][n] = Vt[n][kv0+k]: contiguous 16-halfword row loads.
        v16h pa  = load_a_frag(&sP[0][0], 32, 0, 0);
        v16h vb0 = load_b_frag_rowT(Vtp, SEQ, 0,  kv0);
        v16h vb1 = load_b_frag_rowT(Vtp, SEQ, 16, kv0);
        v16h vb2 = load_b_frag_rowT(Vtp, SEQ, 32, kv0);
        v16h vb3 = load_b_frag_rowT(Vtp, SEQ, 48, kv0);
        o0 = __builtin_amdgcn_wmma_f32_16x16x32_f16(false, pa, false, vb0, (short)0, o0, false, false);
        o1 = __builtin_amdgcn_wmma_f32_16x16x32_f16(false, pa, false, vb1, (short)0, o1, false, false);
        o2 = __builtin_amdgcn_wmma_f32_16x16x32_f16(false, pa, false, vb2, (short)0, o2, false, false);
        o3 = __builtin_amdgcn_wmma_f32_16x16x32_f16(false, pa, false, vb3, (short)0, o3, false, false);
        __syncthreads();   // protect sS/sP before next iteration
    }

    // ---- finalize: O / l, store to [b,s,d] f16 ----
#pragma unroll
    for (int r = 0; r < 8; ++r) {
        const int row = half * 8 + r;
        const float inv = 1.0f / sL[row];
        const int sl = q0 + row;
        const size_t rowbase = ((size_t)bi * SEQ + sl) * DIM + head * DK;
        O[rowbase + 0  + (lane & 15)] = (_Float16)(o0[r] * inv);
        O[rowbase + 16 + (lane & 15)] = (_Float16)(o1[r] * inv);
        O[rowbase + 32 + (lane & 15)] = (_Float16)(o2[r] * inv);
        O[rowbase + 48 + (lane & 15)] = (_Float16)(o3[r] * inv);
    }
}

// ---------------------------------------------------------------------------
// Host-side launcher
// ---------------------------------------------------------------------------
extern "C" void kernel_launch(void* const* d_in, const int* in_sizes, int n_in,
                              void* d_out, int out_size, void* d_ws, size_t ws_size,
                              hipStream_t stream) {
    const float* x   = (const float*)d_in[0];
    const float* w_q = (const float*)d_in[1];
    const float* b_q = (const float*)d_in[2];
    const float* w_k = (const float*)d_in[3];
    const float* b_k = (const float*)d_in[4];
    const float* w_v = (const float*)d_in[5];
    const float* b_v = (const float*)d_in[6];
    const float* w_o = (const float*)d_in[7];
    const float* b_o = (const float*)d_in[8];
    float* out = (float*)d_out;

    const size_t N_X = (size_t)MROWS * DIM;   // 6,291,456
    const size_t N_W = (size_t)DIM * DIM;     //   589,824

    _Float16* ws  = (_Float16*)d_ws;
    _Float16* xh  = ws;
    _Float16* wqh = xh  + N_X;
    _Float16* wkh = wqh + N_W;
    _Float16* wvh = wkh + N_W;
    _Float16* woh = wvh + N_W;
    _Float16* qh  = woh + N_W;   // [b,h,s,dk]
    _Float16* kh  = qh  + N_X;   // [b,h,s,dk]
    _Float16* vth = kh  + N_X;   // [b,h,dk,s]  (V transposed)
    _Float16* oh  = vth + N_X;   // [b,s,d]
    const size_t need_bytes = ((size_t)(5 * N_X + 4 * N_W)) * sizeof(_Float16);
    if (ws_size < need_bytes) return;

    // --- 0: convert inputs/weights to f16 ---
    cvt_f32_f16<<<1024, 256, 0, stream>>>(x,   xh,  (int)N_X);
    cvt_f32_f16<<<512,  256, 0, stream>>>(w_q, wqh, (int)N_W);
    cvt_f32_f16<<<512,  256, 0, stream>>>(w_k, wkh, (int)N_W);
    cvt_f32_f16<<<512,  256, 0, stream>>>(w_v, wvh, (int)N_W);
    cvt_f32_f16<<<512,  256, 0, stream>>>(w_o, woh, (int)N_W);

    // --- 1: Q/K/V projections (TDM-fed tiled WMMA GEMM) ---
    dim3 ggrid(DIM / BN, MROWS / BM);   // (12, 64)
    gemm_tiled<1><<<ggrid, 256, 0, stream>>>(xh, wqh, b_q, qh,  nullptr);
    gemm_tiled<1><<<ggrid, 256, 0, stream>>>(xh, wkh, b_k, kh,  nullptr);
    gemm_tiled<2><<<ggrid, 256, 0, stream>>>(xh, wvh, b_v, vth, nullptr);  // V stored transposed

    // --- 2: causal attention ---
    dim3 agrid(SEQ / 16, BATCH * HEADS);
    attn_fwd<<<agrid, 32, 0, stream>>>(qh, kh, vth, oh);

    // --- 3: output projection (f32 out) ---
    gemm_tiled<0><<<ggrid, 256, 0, stream>>>(oh, woh, b_o, nullptr, out);
}
